// NerfactoModel_55465207661111
// MI455X (gfx1250) — matrix operand order repaired and checked by reading, not compiled, not run on hardware
//
#include <hip/hip_runtime.h>
#include <hip/hip_bf16.h>
#include <math.h>

typedef __attribute__((ext_vector_type(2)))  _Float16 v2h;
typedef __attribute__((ext_vector_type(8)))  _Float16 v8h;
typedef __attribute__((ext_vector_type(16))) _Float16 v16h;
typedef __attribute__((ext_vector_type(8)))  float    v8f;

// resolutions: int(16 * (128^(1/15))^i)
__constant__ int kRes[16] = {16,22,30,42,58,80,111,153,212,294,406,561,776,1072,1482,2048};

#define HMOD 4096
#define P1 1969
#define P2 1941   // 805459861  % 4096

// LDS weight offsets (halves), weights stored TRANSPOSED: wT[o][k] = w[k][o]
#define OFF_DW1 0        // [64][32]
#define OFF_DW2 2048     // [64][64]
#define OFF_DW3 6144     // [16][64]
#define OFF_CW1 7168     // [64][64] (k=63 zero pad)
#define OFF_CW2 11264    // [64][64]
#define OFF_CW3 15360    // [16][64] (o>=3 zero pad)
#define OFF_RW1 16384    // [16][32] (k>=3 zero pad)
#define OFF_RW2 16896    // [16][32] (k>=16 zero pad)
#define W_TOTAL 17408

#define B_DB1 0
#define B_DB2 64
#define B_DB3 128
#define B_CB1 144
#define B_CB2 208
#define B_CB3 272
#define B_RB1 288
#define B_RB2 304
#define B_TOTAL 320

// A-fragment 16x32 f16 from row-major LDS [16][strideH], per ISA 16-bit A layout:
// lanes 0-15: VGPR0-3 = K 0..7, VGPR4-7 = K 16..23; lanes 16-31: K 8..15 / K 24..31
__device__ __forceinline__ v16h lds_frag_a(const _Float16* base, int strideH, int k0, int lane) {
  const int r = lane & 15, sel = lane >> 4;
  const v8h lo = *(const v8h*)(base + r * strideH + k0 + sel * 8);
  const v8h hi = *(const v8h*)(base + r * strideH + k0 + 16 + sel * 8);
  v16h f;
#pragma unroll
  for (int i = 0; i < 8; ++i) { f[i] = lo[i]; f[8 + i] = hi[i]; }
  return f;
}

// B-fragment 32x16 f16 from transposed weights wT[o][k] (row-major [Nout][Kstride]):
// lane n<16: col n0+n, K k0..k0+7 / k0+16..k0+23; lanes 16-31: same col, K k0+8..k0+15 / k0+24..k0+31
__device__ __forceinline__ v16h lds_frag_b(const _Float16* wT, int kStrideH, int n0, int k0, int lane) {
  const int n = lane & 15, kh = (lane >> 4) << 3;
  const v8h* p = (const v8h*)(wT + (n0 + n) * kStrideH + k0 + kh);
  const v8h lo = p[0], hi = p[2];
  v16h f;
#pragma unroll
  for (int i = 0; i < 8; ++i) { f[i] = lo[i]; f[8 + i] = hi[i]; }
  return f;
}

__device__ __forceinline__ v8f wmma_f16(v16h a, v16h b, v8f c) {
  return __builtin_amdgcn_wmma_f32_16x16x32_f16(false, a, false, b, (short)0, c, false, false);
}

// D (16x16 f32: VGPR r -> row mb+r, lane -> col n) -> bias(+relu) -> f16 LDS row-major
__device__ __forceinline__ void store_d(v8f d, _Float16* dst, int strideH, int n0,
                                        const float* biasTile, bool relu, int lane) {
  const int n = lane & 15, mb = (lane >> 4) << 3;
  const float bv = biasTile[n];
#pragma unroll
  for (int r = 0; r < 8; ++r) {
    float v = d[r] + bv;
    if (relu) v = fmaxf(v, 0.f);
    dst[(mb + r) * strideH + n0 + n] = (_Float16)v;
  }
}

// full 64->64 layer: two K-steps x four N-tiles = 8 WMMAs
__device__ __forceinline__ void layer64(const _Float16* in, _Float16* outb,
                                        const _Float16* wT, const float* bias,
                                        bool relu, int lane) {
  v16h a0 = lds_frag_a(in, 64, 0, lane);
  v16h a1 = lds_frag_a(in, 64, 32, lane);
#pragma unroll
  for (int nt = 0; nt < 4; ++nt) {
    v8f c = {};
    c = wmma_f16(a0, lds_frag_b(wT, 64, nt * 16, 0, lane), c);
    c = wmma_f16(a1, lds_frag_b(wT, 64, nt * 16, 32, lane), c);
    store_d(c, outb, 64, nt * 16, bias + nt * 16, relu, lane);
  }
}

__global__ __launch_bounds__(128) void nerfacto_fused(
    const float* __restrict__ ray, const float* __restrict__ dirs,
    const int* __restrict__ cam, const float* __restrict__ aabb,
    const float* __restrict__ ht,
    const float* __restrict__ rw1, const float* __restrict__ rb1,
    const float* __restrict__ rw2, const float* __restrict__ rb2,
    const float* __restrict__ dw1, const float* __restrict__ db1,
    const float* __restrict__ dw2, const float* __restrict__ db2,
    const float* __restrict__ dw3, const float* __restrict__ db3,
    const float* __restrict__ cw1, const float* __restrict__ cb1,
    const float* __restrict__ cw2, const float* __restrict__ cb2,
    const float* __restrict__ cw3, const float* __restrict__ cb3,
    const float* __restrict__ app,
    float* __restrict__ out) {
  __shared__ __align__(16) _Float16 sW[W_TOTAL];
  __shared__ float sB[B_TOTAL];
  __shared__ __align__(16) _Float16 sAct[4][2][16 * 64];

  const int tid = threadIdx.x;
  const int NT = 128;

  // ---- stage all weights to LDS, f16, transposed [Nout][Kpad] ----
  for (int i = tid; i < 64 * 32; i += NT) { int o = i >> 5, k = i & 31; sW[OFF_DW1 + i] = (_Float16)dw1[k * 64 + o]; }
  for (int i = tid; i < 64 * 64; i += NT) { int o = i >> 6, k = i & 63; sW[OFF_DW2 + i] = (_Float16)dw2[k * 64 + o]; }
  for (int i = tid; i < 16 * 64; i += NT) { int o = i >> 6, k = i & 63; sW[OFF_DW3 + i] = (_Float16)dw3[k * 16 + o]; }
  for (int i = tid; i < 64 * 64; i += NT) { int o = i >> 6, k = i & 63; sW[OFF_CW1 + i] = (k < 63) ? (_Float16)cw1[k * 64 + o] : (_Float16)0.f; }
  for (int i = tid; i < 64 * 64; i += NT) { int o = i >> 6, k = i & 63; sW[OFF_CW2 + i] = (_Float16)cw2[k * 64 + o]; }
  for (int i = tid; i < 16 * 64; i += NT) { int o = i >> 6, k = i & 63; sW[OFF_CW3 + i] = (o < 3) ? (_Float16)cw3[k * 3 + o] : (_Float16)0.f; }
  for (int i = tid; i < 16 * 32; i += NT) { int o = i >> 5, k = i & 31; sW[OFF_RW1 + i] = (k < 3)  ? (_Float16)rw1[k * 16 + o] : (_Float16)0.f; }
  for (int i = tid; i < 16 * 32; i += NT) { int o = i >> 5, k = i & 31; sW[OFF_RW2 + i] = (k < 16) ? (_Float16)rw2[k * 16 + o] : (_Float16)0.f; }
  if (tid < 64) {
    sB[B_DB1 + tid] = db1[tid]; sB[B_DB2 + tid] = db2[tid];
    sB[B_CB1 + tid] = cb1[tid]; sB[B_CB2 + tid] = cb2[tid];
  }
  if (tid < 16) {
    sB[B_DB3 + tid] = db3[tid];
    sB[B_RB1 + tid] = rb1[tid]; sB[B_RB2 + tid] = rb2[tid];
    sB[B_CB3 + tid] = (tid < 3) ? cb3[tid] : 0.f;
  }
  __syncthreads();

  const int lane = tid & 31;
  const int wv = tid >> 5;
  _Float16* buf0 = &sAct[wv][0][0];
  _Float16* buf1 = &sAct[wv][1][0];
  const int s0 = (blockIdx.x * 4 + wv) << 4;   // 16 samples per wave

  // ---- hash-grid encoding -> buf0 as A-matrix [16][32] f16 ----
  {
    const int sl = lane & 15;
    const int s = s0 + sl;
    const int lb = (lane >> 4) << 3;           // this lane does 8 levels
    const float a0x = aabb[0], a0y = aabb[1], a0z = aabb[2];
    const float rx = 1.f / (aabb[3] - a0x), ry = 1.f / (aabb[4] - a0y), rz = 1.f / (aabb[5] - a0z);
    const float px = fminf(fmaxf((ray[s * 3 + 0] - a0x) * rx, 0.f), 1.f);
    const float py = fminf(fmaxf((ray[s * 3 + 1] - a0y) * ry, 0.f), 1.f);
    const float pz = fminf(fmaxf((ray[s * 3 + 2] - a0z) * rz, 0.f), 1.f);
#pragma unroll
    for (int j = 0; j < 8; ++j) {
      const int lvl = lb + j;
      const int R = kRes[lvl];
      const float fr = (float)(R - 1);
      int cx = (int)floorf(px * fr); cx = cx < 0 ? 0 : (cx > R - 1 ? R - 1 : cx);
      int cy = (int)floorf(py * fr); cy = cy < 0 ? 0 : (cy > R - 1 ? R - 1 : cy);
      int cz = (int)floorf(pz * fr); cz = cz < 0 ? 0 : (cz > R - 1 ? R - 1 : cz);
      const int h = (cx + cy * P1 + cz * P2) & (HMOD - 1);
      const float* fp = ht + (((size_t)lvl << 12) + h) * 2;
      v2h f; f[0] = (_Float16)fp[0]; f[1] = (_Float16)fp[1];
      *(v2h*)(buf0 + sl * 32 + lvl * 2) = f;
    }
  }

  // ---- density L1: 32 -> 64 (4 WMMAs) ----
  {
    v16h a = lds_frag_a(buf0, 32, 0, lane);
#pragma unroll
    for (int nt = 0; nt < 4; ++nt) {
      v8f c = {};
      c = wmma_f16(a, lds_frag_b(sW + OFF_DW1, 32, nt * 16, 0, lane), c);
      store_d(c, buf1, 64, nt * 16, sB + B_DB1 + nt * 16, true, lane);
    }
  }
  // ---- density L2: 64 -> 64 ----
  layer64(buf1, buf0, sW + OFF_DW2, sB + B_DB2, true, lane);

  // ---- density L3: 64 -> 16; emit density+geo; geo -> cin cols 0..14 ----
  {
    v16h a0 = lds_frag_a(buf0, 64, 0, lane);
    v16h a1 = lds_frag_a(buf0, 64, 32, lane);
    v8f c = {};
    c = wmma_f16(a0, lds_frag_b(sW + OFF_DW3, 64, 0, 0, lane), c);
    c = wmma_f16(a1, lds_frag_b(sW + OFF_DW3, 64, 0, 32, lane), c);
    const int n = lane & 15, mb = (lane >> 4) << 3;
    const float bv = sB[B_DB3 + n];
#pragma unroll
    for (int r = 0; r < 8; ++r) {
      const float v = c[r] + bv;
      const size_t row = (size_t)(s0 + mb + r);
      if (n == 0) {
        out[row * 19 + 3] = fmaxf(v, 0.f);          // density (relu)
      } else {
        out[row * 19 + 3 + n] = v;                  // geo passthrough
        buf1[(mb + r) * 64 + (n - 1)] = (_Float16)v; // geo into color input
      }
    }
  }

  // ---- direction MLP: 3 -> 16 -> 16, result -> cin cols 15..30 ----
  {
    v8h z = {};
    *(v8h*)(buf0 + lane * 16) = z;       // zero 16x32 dir A region
    *(v8h*)(buf0 + lane * 16 + 8) = z;   // (wait/ordering handled per-wave, LDS in-order)
    if (lane < 16) {
      const float* dp = dirs + (size_t)(s0 + lane) * 3;
      buf0[lane * 32 + 0] = (_Float16)dp[0];
      buf0[lane * 32 + 1] = (_Float16)dp[1];
      buf0[lane * 32 + 2] = (_Float16)dp[2];
    }
    v16h a = lds_frag_a(buf0, 32, 0, lane);
    v8f c = {};
    c = wmma_f16(a, lds_frag_b(sW + OFF_RW1, 32, 0, 0, lane), c);
    store_d(c, buf0, 32, 0, sB + B_RB1, true, lane);  // hidden back to cols 0..15 (16..31 stay 0)
    v16h a2 = lds_frag_a(buf0, 32, 0, lane);
    v8f c2 = {};
    c2 = wmma_f16(a2, lds_frag_b(sW + OFF_RW2, 32, 0, 0, lane), c2);
    store_d(c2, buf1, 64, 15, sB + B_RB2, false, lane);
  }

  // ---- appearance embedding -> cin cols 31..62, zero col 63 ----
  {
    const int sl = lane & 15;
    const int ci = cam[s0 + sl];
    const int fh = (lane >> 4) << 4;
    const float4* ap = (const float4*)(app + ((size_t)ci << 5) + fh);
    _Float16* dst = buf1 + sl * 64 + 31 + fh;
#pragma unroll
    for (int q = 0; q < 4; ++q) {
      const float4 v = ap[q];
      dst[q * 4 + 0] = (_Float16)v.x; dst[q * 4 + 1] = (_Float16)v.y;
      dst[q * 4 + 2] = (_Float16)v.z; dst[q * 4 + 3] = (_Float16)v.w;
    }
    if (lane < 16) buf1[lane * 64 + 63] = (_Float16)0.f;
  }

  // ---- color MLP: 64 -> 64 -> 64 -> 3(sigmoid) ----
  layer64(buf1, buf0, sW + OFF_CW1, sB + B_CB1, true, lane);
  layer64(buf0, buf1, sW + OFF_CW2, sB + B_CB2, true, lane);
  {
    v16h a0 = lds_frag_a(buf1, 64, 0, lane);
    v16h a1 = lds_frag_a(buf1, 64, 32, lane);
    v8f c = {};
    c = wmma_f16(a0, lds_frag_b(sW + OFF_CW3, 64, 0, 0, lane), c);
    c = wmma_f16(a1, lds_frag_b(sW + OFF_CW3, 64, 0, 32, lane), c);
    const int n = lane & 15, mb = (lane >> 4) << 3;
    if (n < 3) {
      const float bv = sB[B_CB3 + n];
#pragma unroll
      for (int r = 0; r < 8; ++r) {
        const float v = c[r] + bv;
        out[(size_t)(s0 + mb + r) * 19 + n] = 1.f / (1.f + __expf(-v));
      }
    }
  }
}

extern "C" void kernel_launch(void* const* d_in, const int* in_sizes, int n_in,
                              void* d_out, int out_size, void* d_ws, size_t ws_size,
                              hipStream_t stream) {
  (void)n_in; (void)out_size; (void)d_ws; (void)ws_size;
  const float* ray  = (const float*)d_in[0];
  const float* dirs = (const float*)d_in[1];
  const int*   cam  = (const int*)d_in[2];
  const float* aabb = (const float*)d_in[3];
  const float* ht   = (const float*)d_in[4];
  const float* rw1 = (const float*)d_in[5];  const float* rb1 = (const float*)d_in[6];
  const float* rw2 = (const float*)d_in[7];  const float* rb2 = (const float*)d_in[8];
  const float* dw1 = (const float*)d_in[9];  const float* db1 = (const float*)d_in[10];
  const float* dw2 = (const float*)d_in[11]; const float* db2 = (const float*)d_in[12];
  const float* dw3 = (const float*)d_in[13]; const float* db3 = (const float*)d_in[14];
  const float* cw1 = (const float*)d_in[15]; const float* cb1 = (const float*)d_in[16];
  const float* cw2 = (const float*)d_in[17]; const float* cb2 = (const float*)d_in[18];
  const float* cw3 = (const float*)d_in[19]; const float* cb3 = (const float*)d_in[20];
  const float* app = (const float*)d_in[21];
  float* out = (float*)d_out;

  const int n = in_sizes[2];        // N samples (camera_indices is (N,))
  const int blocks = n / 64;        // 4 waves/block * 16 samples/wave; N=2M divides evenly

  nerfacto_fused<<<blocks, 128, 0, stream>>>(
      ray, dirs, cam, aabb, ht,
      rw1, rb1, rw2, rb2,
      dw1, db1, dw2, db2, dw3, db3,
      cw1, cb1, cw2, cb2, cw3, cb3,
      app, out);
}
